// CombinedAttention_LN_35983236006609
// MI455X (gfx1250) — compile-verified, matrix operand order checked by reading, not run on hardware
//
#include <hip/hip_runtime.h>
#include <hip/hip_bf16.h>

typedef __attribute__((ext_vector_type(16))) _Float16 v16h;
typedef __attribute__((ext_vector_type(8)))  _Float16 v8h;
typedef __attribute__((ext_vector_type(8)))  float    v8f;

#define BATCH 2
#define SEQ   2048
#define DIM   512
#define HEADS 8
#define HDIM  64
#define BN    (BATCH * SEQ)   // 4096 tokens
#define TDIM  (3 * DIM)       // 1536

// ---------------------------------------------------------------------------
// WMMA helper (CDNA5 gfx1250, wave32)
// ---------------------------------------------------------------------------
__device__ inline v8f wmma_f16(v16h a, v16h b, v8f c) {
  return __builtin_amdgcn_wmma_f32_16x16x32_f16(
      /*neg_a=*/false, a, /*neg_b=*/false, b,
      /*c_mod=*/(short)0, c, /*reuse_a=*/false, /*reuse_b=*/false);
}

// Fragment loader: 16 rows x 32 K from row-major storage (ld in halves).
// ISA layout: lane<16 holds row=lane, K={0..7,16..23}; lane>=16 K={8..15,24..31}.
// Serves as A-fragment, and as B-fragment for N-major operands.
__device__ inline v16h load_frag_rowmajor(const _Float16* base, int ld) {
  const int lane = threadIdx.x & 31;
  const _Float16* p = base + (size_t)(lane & 15) * ld + ((lane >> 4) << 3);
  v8h lo = *(const v8h*)p;          // K = khi .. khi+7
  v8h hi = *(const v8h*)(p + 16);   // K = khi+16 .. khi+23
  v16h f;
#pragma unroll
  for (int i = 0; i < 8; ++i) { f[i] = lo[i]; f[i + 8] = hi[i]; }
  return f;
}

// Async global -> LDS 128-bit copy (gfx1250, ASYNCcnt tracked, in-order).
__device__ inline void async_copy_b128(const void* gptr, void* lptr) {
  uint32_t lds = (uint32_t)(uintptr_t)lptr;    // low 32 bits == LDS byte offset
  uint64_t ga  = (uint64_t)(uintptr_t)gptr;
  asm volatile("global_load_async_to_lds_b128 %0, %1, off"
               :: "v"(lds), "v"(ga) : "memory");
}
#define ASYNC_WAIT(n) asm volatile("s_wait_asynccnt " #n ::: "memory")

// ---------------------------------------------------------------------------
// 1a) f32 -> f16 cast (activations)
// ---------------------------------------------------------------------------
__global__ void cast_f16_kernel(const float* __restrict__ in,
                                _Float16* __restrict__ out, int count) {
  int i = blockIdx.x * blockDim.x + threadIdx.x;
  if (i < count) out[i] = (_Float16)in[i];
}

// 1b) f32 [K][N] -> f16 [N][K] cast+transpose (weights)
__global__ void cast_transpose_kernel(const float* __restrict__ in,
                                      _Float16* __restrict__ out, int K, int N) {
  int i = blockIdx.x * blockDim.x + threadIdx.x;
  if (i < K * N) {
    int k = i / N, n = i - k * N;
    out[(size_t)n * K + k] = (_Float16)in[i];
  }
}

// ---------------------------------------------------------------------------
// 2/5) WMMA GEMM: C[M,N] = A[M,K](f16) * Bt[N,K](f16)^T + bias[N]  (f32 out)
//     128x64 block tile, 4 waves (wave w: rows w*32..w*32+31), double-buffered
//     LDS staging via global_load_async_to_lds_b128 (6 copies/thread/tile).
// ---------------------------------------------------------------------------
__global__ __launch_bounds__(128)
void gemm_wmma_kernel(const _Float16* __restrict__ A,
                      const _Float16* __restrict__ Bt,
                      const float* __restrict__ bias,
                      float* __restrict__ C,
                      int Mdim, int Ndim, int Kdim) {
  __shared__ __align__(16) _Float16 Alds[2][128][32];
  __shared__ __align__(16) _Float16 Blds[2][64][32];

  const int wave = threadIdx.x >> 5;
  const int lane = threadIdx.x & 31;
  const int m0 = blockIdx.y * 128;
  const int n0 = blockIdx.x * 64;

  v8f zero = {};
  v8f acc[2][4];
#pragma unroll
  for (int h = 0; h < 2; ++h)
#pragma unroll
    for (int t = 0; t < 4; ++t) acc[h][t] = zero;

  // stage tile at K-offset k0 into buffer p: A 128x32 (4 b128/thr), B 64x32 (2)
  auto stage = [&](int p, int k0) {
#pragma unroll
    for (int u = 0; u < 4; ++u) {
      int q = threadIdx.x * 4 + u;
      int rr = q >> 2, pp = (q & 3) * 8;
      async_copy_b128(A + (size_t)(m0 + rr) * Kdim + k0 + pp, &Alds[p][rr][pp]);
    }
#pragma unroll
    for (int u = 0; u < 2; ++u) {
      int q = threadIdx.x * 2 + u;
      int rr = q >> 2, pp = (q & 3) * 8;
      async_copy_b128(Bt + (size_t)(n0 + rr) * Kdim + k0 + pp, &Blds[p][rr][pp]);
    }
  };

  stage(0, 0);
  int p = 0;
  for (int k0 = 0; k0 < Kdim; k0 += 32) {
    if (k0 + 32 < Kdim) {
      stage(p ^ 1, k0 + 32);   // prefetch next tile into other buffer
      ASYNC_WAIT(6);           // 6 outstanding = next tile; current tile retired
    } else {
      ASYNC_WAIT(0);
    }
    __syncthreads();           // current buffer visible to all waves

    v16h af0 = load_frag_rowmajor(&Alds[p][wave * 32 +  0][0], 32);
    v16h af1 = load_frag_rowmajor(&Alds[p][wave * 32 + 16][0], 32);
#pragma unroll
    for (int t = 0; t < 4; ++t) {
      v16h bf = load_frag_rowmajor(&Blds[p][t * 16][0], 32);
      acc[0][t] = wmma_f16(af0, bf, acc[0][t]);
      acc[1][t] = wmma_f16(af1, bf, acc[1][t]);
    }
    __syncthreads();           // all reads done before this buffer is re-staged
    p ^= 1;
  }

  // Epilogue: C element (v,lane) -> row = v + 8*(lane>=16), col = lane&15
  const int col = lane & 15;
  const int mhi = (lane >> 4) * 8;
#pragma unroll
  for (int h = 0; h < 2; ++h)
#pragma unroll
    for (int t = 0; t < 4; ++t) {
      int cn = n0 + t * 16 + col;
      float bv = bias[cn];
#pragma unroll
      for (int v = 0; v < 8; ++v) {
        int row = m0 + wave * 32 + h * 16 + mhi + v;
        C[(size_t)row * Ndim + cn] = acc[h][t][v] + bv;
      }
    }
}

// ---------------------------------------------------------------------------
// 3) Per-head LayerNorm on Q,K + pack. Q/K -> [b,h,n,64] f16 (Q scaled 1/8);
//    V -> TRANSPOSED [b,h,64,n] f16 so the P*V B-fragment is row-major.
// ---------------------------------------------------------------------------
__global__ __launch_bounds__(256)
void ln_pack_kernel(const float* __restrict__ qkv,
                    const float* __restrict__ qn_w, const float* __restrict__ qn_b,
                    const float* __restrict__ kn_w, const float* __restrict__ kn_b,
                    _Float16* __restrict__ Q, _Float16* __restrict__ K,
                    _Float16* __restrict__ Vt) {
  const int task = blockIdx.x * 8 + (threadIdx.x >> 5);   // token*8 + head
  const int lane = threadIdx.x & 31;
  const int head = task & 7;
  const int tok  = task >> 3;             // b*SEQ + n
  const int b    = tok >> 11;
  const int n    = tok & (SEQ - 1);

  const float* row = qkv + (size_t)tok * TDIM;
  const int d0 = lane * 2;
  const size_t bh = (size_t)(b * HEADS + head);
  const size_t outoff = (bh * SEQ + n) * HDIM;

  // ---- Q: layernorm, * qn_w + qn_b, * 1/sqrt(64) ----
  {
    float x0 = row[0 * DIM + head * HDIM + d0];
    float x1 = row[0 * DIM + head * HDIM + d0 + 1];
    float s = x0 + x1;
#pragma unroll
    for (int m = 16; m >= 1; m >>= 1) s += __shfl_xor(s, m, 32);
    float mu = s * (1.0f / 64.0f);
    float e0 = x0 - mu, e1 = x1 - mu;
    float vs = e0 * e0 + e1 * e1;
#pragma unroll
    for (int m = 16; m >= 1; m >>= 1) vs += __shfl_xor(vs, m, 32);
    float r = rsqrtf(vs * (1.0f / 64.0f) + 1e-5f);
    Q[outoff + d0]     = (_Float16)((e0 * r * qn_w[d0]     + qn_b[d0])     * 0.125f);
    Q[outoff + d0 + 1] = (_Float16)((e1 * r * qn_w[d0 + 1] + qn_b[d0 + 1]) * 0.125f);
  }
  // ---- K: layernorm, * kn_w + kn_b ----
  {
    float x0 = row[1 * DIM + head * HDIM + d0];
    float x1 = row[1 * DIM + head * HDIM + d0 + 1];
    float s = x0 + x1;
#pragma unroll
    for (int m = 16; m >= 1; m >>= 1) s += __shfl_xor(s, m, 32);
    float mu = s * (1.0f / 64.0f);
    float e0 = x0 - mu, e1 = x1 - mu;
    float vs = e0 * e0 + e1 * e1;
#pragma unroll
    for (int m = 16; m >= 1; m >>= 1) vs += __shfl_xor(vs, m, 32);
    float r = rsqrtf(vs * (1.0f / 64.0f) + 1e-5f);
    K[outoff + d0]     = (_Float16)(e0 * r * kn_w[d0]     + kn_b[d0]);
    K[outoff + d0 + 1] = (_Float16)(e1 * r * kn_w[d0 + 1] + kn_b[d0 + 1]);
  }
  // ---- V: transposed cast: Vt[bh][d][n] ----
  {
    const size_t vbase = bh * HDIM * SEQ + n;
    Vt[vbase + (size_t)d0 * SEQ]       = (_Float16)row[2 * DIM + head * HDIM + d0];
    Vt[vbase + (size_t)(d0 + 1) * SEQ] = (_Float16)row[2 * DIM + head * HDIM + d0 + 1];
  }
}

// ---------------------------------------------------------------------------
// 4) Flash attention with analytic periodic positional bias.
//    grid = (SEQ/64 q-tiles, BATCH*HEADS); 4 waves x 16 query rows.
//    K/V 64x64 tiles staged ONCE per block (async, double-buffered) instead of
//    4x redundant per-wave global fragment loads.
// ---------------------------------------------------------------------------
__global__ __launch_bounds__(128)
void flash_attn_kernel(const _Float16* __restrict__ Q,
                       const _Float16* __restrict__ K,
                       const _Float16* __restrict__ Vt,
                       const float* __restrict__ pos,
                       _Float16* __restrict__ O) {
  const int bh   = blockIdx.y;
  const int wave = threadIdx.x >> 5;
  const int lane = threadIdx.x & 31;
  const int col  = lane & 15;
  const int hi   = lane >> 4;
  const int qrow0 = blockIdx.x * 64 + wave * 16;

  const _Float16* Qb  = Q  + (size_t)bh * SEQ * HDIM;
  const _Float16* Kb  = K  + (size_t)bh * SEQ * HDIM;
  const _Float16* Vtb = Vt + (size_t)bh * HDIM * SEQ;

  __shared__ __align__(16) _Float16 Klds[2][64][64];   // [key][d]
  __shared__ __align__(16) _Float16 Vlds[2][64][64];   // [d][key]
  __shared__ __align__(16) _Float16 lds_p[4][16][72];  // wave-private P tiles

  v16h qf0 = load_frag_rowmajor(Qb + (size_t)qrow0 * HDIM + 0,  HDIM);
  v16h qf1 = load_frag_rowmajor(Qb + (size_t)qrow0 * HDIM + 32, HDIM);

  v8f zero = {};
  v8f oacc[4];
#pragma unroll
  for (int t = 0; t < 4; ++t) oacc[t] = zero;

  float mrow[8], lrow[8], pix[8], piy[8];
#pragma unroll
  for (int v = 0; v < 8; ++v) {
    mrow[v] = -1e30f; lrow[v] = 0.0f;
    int r = qrow0 + hi * 8 + v;
    pix[v] = pos[2 * r];
    piy[v] = pos[2 * r + 1];
  }

  // stage 64-key K and V tiles for key offset j0 into buffer p (8 b128/thread)
  auto stage = [&](int p, int j0) {
#pragma unroll
    for (int u = 0; u < 4; ++u) {
      int q = threadIdx.x * 4 + u;
      int rr = q >> 3, pp = (q & 7) * 8;
      async_copy_b128(Kb + (size_t)(j0 + rr) * HDIM + pp, &Klds[p][rr][pp]);
      async_copy_b128(Vtb + (size_t)rr * SEQ + j0 + pp,   &Vlds[p][rr][pp]);
    }
  };

  stage(0, 0);
  int p = 0;
  for (int j0 = 0; j0 < SEQ; j0 += 64) {
    if (j0 + 64 < SEQ) {
      stage(p ^ 1, j0 + 64);
      ASYNC_WAIT(8);           // 8 outstanding = next tile; current tile retired
    } else {
      ASYNC_WAIT(0);
    }
    __syncthreads();           // current K/V buffer visible to all waves

    // ---- S = Q K^T + bias : four 16x16 tiles covering 64 keys ----
    v8f s[4];
#pragma unroll
    for (int t = 0; t < 4; ++t) {
      v16h kf0 = load_frag_rowmajor(&Klds[p][t * 16][0],  64);
      v16h kf1 = load_frag_rowmajor(&Klds[p][t * 16][32], 64);
      v8f a = zero;
      a = wmma_f16(qf0, kf0, a);
      a = wmma_f16(qf1, kf1, a);
      int jc = j0 + t * 16 + col;
      float pjx = pos[2 * jc], pjy = pos[2 * jc + 1];
#pragma unroll
      for (int v = 0; v < 8; ++v) {
        float dx = fabsf(pix[v] - pjx); dx = fmodf(dx + 0.5f, 1.0f) - 0.5f;
        float dy = fabsf(piy[v] - pjy); dy = fmodf(dy + 0.5f, 1.0f) - 0.5f;
        a[v] -= dx * dx + dy * dy;            // bias = -edist
      }
      s[t] = a;
    }

    // ---- streaming softmax update (rows live in 16-lane half-waves) ----
    float newm[8], scal[8];
#pragma unroll
    for (int v = 0; v < 8; ++v) {
      float m = fmaxf(fmaxf(s[0][v], s[1][v]), fmaxf(s[2][v], s[3][v]));
#pragma unroll
      for (int off = 8; off >= 1; off >>= 1) m = fmaxf(m, __shfl_xor(m, off, 32));
      newm[v] = fmaxf(mrow[v], m);
      scal[v] = __expf(mrow[v] - newm[v]);
      mrow[v] = newm[v];
    }
#pragma unroll
    for (int v = 0; v < 8; ++v) {
      float rs = 0.0f;
#pragma unroll
      for (int t = 0; t < 4; ++t) {
        float e = __expf(s[t][v] - newm[v]);
        s[t][v] = e;
        rs += e;
      }
#pragma unroll
      for (int off = 8; off >= 1; off >>= 1) rs += __shfl_xor(rs, off, 32);
      lrow[v] = lrow[v] * scal[v] + rs;
#pragma unroll
      for (int t = 0; t < 4; ++t) oacc[t][v] *= scal[v];
    }

    // ---- C-layout P -> A-fragment layout via wave-private LDS tile ----
    // (DS ops are in-order per wave; compiler inserts s_wait_dscnt for the RAW)
#pragma unroll
    for (int t = 0; t < 4; ++t)
#pragma unroll
      for (int v = 0; v < 8; ++v)
        lds_p[wave][hi * 8 + v][t * 16 + col] = (_Float16)s[t][v];

    v16h pf0 = load_frag_rowmajor(&lds_p[wave][0][0],  72);  // keys j0..j0+31
    v16h pf1 = load_frag_rowmajor(&lds_p[wave][0][32], 72);  // keys j0+32..j0+63

#pragma unroll
    for (int t = 0; t < 4; ++t) {
      v16h vf0 = load_frag_rowmajor(&Vlds[p][t * 16][0],  64);
      v16h vf1 = load_frag_rowmajor(&Vlds[p][t * 16][32], 64);
      oacc[t] = wmma_f16(pf0, vf0, oacc[t]);
      oacc[t] = wmma_f16(pf1, vf1, oacc[t]);
    }
    __syncthreads();           // all reads done before this buffer is re-staged
    p ^= 1;
  }

  // ---- finalize: O[b][n][h*64+d] = oacc / l  (f16 for the out-proj GEMM) ----
  const int b = bh / HEADS, h = bh % HEADS;
#pragma unroll
  for (int v = 0; v < 8; ++v) {
    float inv = 1.0f / lrow[v];
    int r = qrow0 + hi * 8 + v;
    _Float16* op = O + ((size_t)b * SEQ + r) * DIM + h * HDIM;
#pragma unroll
    for (int t = 0; t < 4; ++t)
      op[t * 16 + col] = (_Float16)(oacc[t][v] * inv);
  }
}

// ---------------------------------------------------------------------------
// Host-side orchestration
// ---------------------------------------------------------------------------
extern "C" void kernel_launch(void* const* d_in, const int* in_sizes, int n_in,
                              void* d_out, int out_size, void* d_ws, size_t ws_size,
                              hipStream_t stream) {
  const float* x     = (const float*)d_in[0];
  const float* pos   = (const float*)d_in[1];
  const float* W_in  = (const float*)d_in[2];
  const float* b_in  = (const float*)d_in[3];
  const float* qn_w  = (const float*)d_in[4];
  const float* qn_b  = (const float*)d_in[5];
  const float* kn_w  = (const float*)d_in[6];
  const float* kn_b  = (const float*)d_in[7];
  const float* W_out = (const float*)d_in[8];
  const float* b_out = (const float*)d_in[9];
  float* out = (float*)d_out;

  // Workspace carve-up (~48 MB total)
  char* w = (char*)d_ws;
  _Float16* Xh     = (_Float16*)w;                w += (size_t)BN * DIM * 2;
  _Float16* WinT   = (_Float16*)w;                w += (size_t)DIM * TDIM * 2;  // [1536][512]
  _Float16* WoutT  = (_Float16*)w;                w += (size_t)DIM * DIM * 2;   // [512][512]
  float*    qkv    = (float*)w;                   w += (size_t)BN * TDIM * 4;
  _Float16* Qh     = (_Float16*)w;                w += (size_t)BN * DIM * 2;
  _Float16* Kh     = (_Float16*)w;                w += (size_t)BN * DIM * 2;
  _Float16* Vth    = (_Float16*)w;                w += (size_t)BN * DIM * 2;
  _Float16* Ah     = (_Float16*)w;                w += (size_t)BN * DIM * 2;

  // 1) casts (weights transposed to [N][K])
  {
    int c1 = BN * DIM, c2 = DIM * TDIM, c3 = DIM * DIM;
    cast_f16_kernel<<<(c1 + 255) / 256, 256, 0, stream>>>(x, Xh, c1);
    cast_transpose_kernel<<<(c2 + 255) / 256, 256, 0, stream>>>(W_in, WinT, DIM, TDIM);
    cast_transpose_kernel<<<(c3 + 255) / 256, 256, 0, stream>>>(W_out, WoutT, DIM, DIM);
  }
  // 2) QKV projection: qkv[BN,1536] = Xh * WinT^T + b_in
  gemm_wmma_kernel<<<dim3(TDIM / 64, BN / 128), 128, 0, stream>>>(
      Xh, WinT, b_in, qkv, BN, TDIM, DIM);
  // 3) layernorm + pack (V transposed)
  ln_pack_kernel<<<BN * HEADS / 8, 256, 0, stream>>>(
      qkv, qn_w, qn_b, kn_w, kn_b, Qh, Kh, Vth);
  // 4) flash attention
  flash_attn_kernel<<<dim3(SEQ / 64, BATCH * HEADS), 128, 0, stream>>>(
      Qh, Kh, Vth, pos, Ah);
  // 5) output projection: out[BN,512] = Ah * WoutT^T + b_out
  gemm_wmma_kernel<<<dim3(DIM / 64, BN / 128), 128, 0, stream>>>(
      Ah, WoutT, b_out, out, BN, DIM, DIM);
}
